// SAKELayer_75531294868026
// MI455X (gfx1250) — matrix operand order repaired and checked by reading, not compiled, not run on hardware
//
#include <hip/hip_runtime.h>
#include <hip/hip_bf16.h>

// ---------------------------------------------------------------------------
// SAKE layer forward, fused, for gfx1250 (MI455X).
//   N=1024, IN=H=OUT=64, C=32.  f32 in/out, bf16 WMMA (f32 accumulate).
// ---------------------------------------------------------------------------

#define NPTS 1024
#define EPSV 1e-5f
#define INFV 1e5f

typedef __attribute__((ext_vector_type(16))) __bf16       v16bf;
typedef __attribute__((ext_vector_type(8)))  float        v8f;
typedef __attribute__((ext_vector_type(8)))  unsigned int v8u;
typedef __attribute__((ext_vector_type(4)))  unsigned int v4u;

union U32B { v8u u;    v16bf b; };   // 32-byte frag view
union U16B { v4u u[2]; v16bf b; };

__device__ __forceinline__ void wave_fence() {
    __builtin_amdgcn_wave_barrier();
    asm volatile("s_wait_dscnt 0" ::: "memory");   // CDNA5 split DS counter
    __builtin_amdgcn_wave_barrier();
}

__device__ __forceinline__ float silu_f(float v) { return v / (1.f + __expf(-v)); }

__device__ __forceinline__ v8f wmma_bf16(v16bf a, v16bf b, v8f c) {
    return __builtin_amdgcn_wmma_f32_16x16x32_bf16(false, a, false, b, (short)0, c,
                                                   false, false);
}

__device__ __forceinline__ v8f splat8(float b) {
    v8f c;
#pragma unroll
    for (int v = 0; v < 8; ++v) c[v] = b;
    return c;
}

// B fragment from LDS weight tile stored as Wt[n][k] bf16 (row stride 64).
__device__ __forceinline__ v16bf ldW(const __bf16* W, int l15, int half, int kt, int nt) {
    const __bf16* p = W + (nt * 16 + l15) * 64 + kt * 32 + half * 8;
    U16B u; u.u[0] = *(const v4u*)p; u.u[1] = *(const v4u*)(p + 16); return u.b;
}

// A fragment from a bf16 [16][64] row-major LDS tile.
__device__ __forceinline__ v16bf ldA(const __bf16* t, int l15, int half, int kt) {
    const __bf16* p = t + l15 * 64 + kt * 32 + half * 8;
    U16B u; u.u[0] = *(const v4u*)p; u.u[1] = *(const v4u*)(p + 16); return u.b;
}

// C fragment -> bf16 [16][64] LDS tile.
__device__ __forceinline__ void stC(__bf16* t, int l15, int half, int g, v8f c) {
#pragma unroll
    for (int v = 0; v < 8; ++v)
        t[(v + 8 * half) * 64 + g * 16 + l15] = (__bf16)c[v];
}

struct alignas(64) WaveScratch {
    float  stage[16 * 64];   // 4KB: f32 B-row tile / bf16 round-trip tile
    float  Ai[64];
    float  uu[64];
    float  w129[64];
    float  comb[16];
    float  phiacc[16];
    float  pad0[32];
    float4 dirs[16];
    float  nodein[192];
    float  sq[32];
    float  tmp[64];
};

// hidden = SiLU(A_i + B_j + norm*w129) built directly in A-fragment layout;
// also returns per-lane partial of hidden . u  (semantic attention logit).
template <bool MF>
__device__ __forceinline__ float build_hidden(WaveScratch* w, int l15, int half,
                                              float nrm, v16bf* af) {
    float dot = 0.f;
    const float* Ai = w->Ai;
    const float* uu = w->uu;
    const float* w9 = w->w129;
    const float* br = w->stage + l15 * 64;
#pragma unroll
    for (int kt = 0; kt < 2; ++kt) {
        v16bf a;
#pragma unroll
        for (int c = 0; c < 2; ++c) {
            int kb = kt * 32 + half * 8 + c * 16;
            v8f fa = *(const v8f*)(Ai + kb);
            v8f fb = *(const v8f*)(br + kb);
            v8f fw = *(const v8f*)(w9 + kb);
            v8f fu = *(const v8f*)(uu + kb);
#pragma unroll
            for (int e = 0; e < 8; ++e) {
                float pre = fa[e] + fb[e] + nrm * fw[e];
                float hid = silu_f(pre);
                dot += hid * fu[e];
                if (MF) a[c * 8 + e] = (__bf16)hid;
            }
        }
        if (MF) af[kt] = a;
    }
    return dot;
}

__device__ __forceinline__ void stageB(WaveScratch* w, const float* Bsrc, int lane) {
    const float4* s = (const float4*)Bsrc;
    float4*       d = (float4*)w->stage;
#pragma unroll
    for (int q = 0; q < 8; ++q) d[q * 32 + lane] = s[q * 32 + lane];
}

// ------------------------------- prep kernels -------------------------------

__global__ void sake_prep_rows(const float* __restrict__ h, const float* __restrict__ ew1,
                               const float* __restrict__ eb1, float* __restrict__ A,
                               float* __restrict__ B) {
    __shared__ float sh[64];
    int i = blockIdx.x, t = threadIdx.x;
    sh[t] = h[i * 64 + t];
    __syncthreads();
    float a = eb1[t], b = 0.f;
#pragma unroll 8
    for (int k = 0; k < 64; ++k) {
        float hv = sh[k];
        a += hv * ew1[k * 64 + t];
        b += hv * ew1[(64 + k) * 64 + t];
    }
    A[i * 64 + t] = a;
    B[i * 64 + t] = b;
}

__global__ void sake_prep_u(const float* __restrict__ ew2, const float* __restrict__ eb2,
                            const float* __restrict__ aw, const float* __restrict__ ab,
                            float* __restrict__ U, float* __restrict__ C0) {
    int t = threadIdx.x;
    float u = 0.f;
#pragma unroll 8
    for (int q = 0; q < 64; ++q) u += ew2[t * 64 + q] * aw[q];
    U[t] = u;
    if (t == 0) {
        float c = ab[0];
        for (int q = 0; q < 64; ++q) c += eb2[q] * aw[q];
        C0[0] = c;
    }
}

// -------------------------------- main kernel --------------------------------

__global__ __launch_bounds__(128, 1) void sake_main(
    const float* __restrict__ h, const float* __restrict__ x,
    const float* __restrict__ ew1, const float* __restrict__ ew2,
    const float* __restrict__ eb2, const float* __restrict__ nw1,
    const float* __restrict__ nb1, const float* __restrict__ nw2,
    const float* __restrict__ nb2, const float* __restrict__ cw1,
    const float* __restrict__ cb1, const float* __restrict__ cw2,
    const float* __restrict__ fw1, const float* __restrict__ fb1,
    const float* __restrict__ fw2, const float* __restrict__ fb2,
    const float* __restrict__ pw1, const float* __restrict__ pb1,
    const float* __restrict__ pw2, const float* __restrict__ pb2,
    const float* __restrict__ log_gamma, const float* __restrict__ Arow,
    const float* __restrict__ Brow, const float* __restrict__ Uvec,
    const float* __restrict__ C0p, float* __restrict__ out) {
    __shared__ alignas(32) __bf16 sEw2[4096];
    __shared__ alignas(32) __bf16 sCw1[4096];
    __shared__ alignas(32) __bf16 sFw1[4096];
    __shared__ alignas(32) __bf16 sFw2[2048];
    __shared__ WaveScratch sw[4];

    const int tid = threadIdx.x;
    // weights -> LDS as Wt[n][k] bf16 (B-fragment friendly)
    for (int t = tid; t < 4096; t += 128) {
        int n = t >> 6, k = t & 63;
        sEw2[t] = (__bf16)ew2[k * 64 + n];
        sCw1[t] = (__bf16)cw1[k * 64 + n];
        sFw1[t] = (__bf16)fw1[k * 64 + n];
    }
    for (int t = tid; t < 2048; t += 128) {
        int n = t >> 6, k = t & 63;
        sFw2[t] = (__bf16)fw2[k * 32 + n];
    }
    __syncthreads();

    const int wv = tid >> 5, lane = tid & 31, half = lane >> 4, l15 = lane & 15;
    WaveScratch* w = &sw[wv];
    const int i = blockIdx.x * 4 + wv;

    w->Ai[lane]        = Arow[i * 64 + lane];
    w->Ai[lane + 32]   = Arow[i * 64 + lane + 32];
    w->uu[lane]        = Uvec[lane];
    w->uu[lane + 32]   = Uvec[lane + 32];
    w->w129[lane]      = ew1[128 * 64 + lane];
    w->w129[lane + 32] = ew1[128 * 64 + lane + 32];
    wave_fence();

    const float gamma = __expf(log_gamma[0]);
    const float c0    = C0p[0];
    const float xi0 = x[3 * i], xi1 = x[3 * i + 1], xi2 = x[3 * i + 2];

    float eb2v[4], fb1v[4], cb1v[4], cw2v[4], fb2v[2];
#pragma unroll
    for (int g = 0; g < 4; ++g) {
        eb2v[g] = eb2[g * 16 + l15];
        fb1v[g] = fb1[g * 16 + l15];
        cb1v[g] = cb1[g * 16 + l15];
        cw2v[g] = cw2[g * 16 + l15];
    }
    fb2v[0] = fb2[l15];
    fb2v[1] = fb2[16 + l15];

    // ---------------- pass 1: online softmax statistics over all j ----------
    float m_s = -1e30f, sg_s = 0.f, sg_a = 0.f, Tt = 0.f;
    for (int jt = 0; jt < 64; ++jt) {
        stageB(w, Brow + jt * 1024, lane);
        wave_fence();
        const int j = jt * 16 + l15;
        const float* xp = x + 3 * j;
        float dx0 = xi0 - xp[0], dx1 = xi1 - xp[1], dx2 = xi2 - xp[2];
        float nrm = sqrtf(dx0 * dx0 + dx1 * dx1 + dx2 * dx2 + EPSV);
        float dot = build_hidden<false>(w, l15, half, nrm, nullptr);
        dot += __shfl_xor(dot, 16, 32);
        const bool dg = (j == i);
        float sl = dot + c0;
        float s  = (sl > 0.f ? sl : 0.01f * sl) - (dg ? INFV : 0.f);
        float a  = -(nrm + (dg ? INFV : 0.f)) * gamma;
        sg_a += __expf(a);
        float mn = fmaxf(m_s, s);
        float sc = __expf(m_s - mn);
        sg_s = sg_s * sc + __expf(s - mn);
        Tt   = Tt * sc + __expf(a + s - mn);
        m_s  = mn;
    }
    // merge per-lane stats across the 16-lane group (both halves identical)
#pragma unroll
    for (int mk = 1; mk < 16; mk <<= 1) {
        float mo = __shfl_xor(m_s, mk, 32);
        float so = __shfl_xor(sg_s, mk, 32);
        float To = __shfl_xor(Tt, mk, 32);
        float ao = __shfl_xor(sg_a, mk, 32);
        float mn = fmaxf(m_s, mo);
        float e1 = __expf(m_s - mn), e2 = __expf(mo - mn);
        sg_s = sg_s * e1 + so * e2;
        Tt   = Tt * e1 + To * e2;
        sg_a += ao;
        m_s = mn;
    }
    const float invT = 1.f / (Tt + EPSV * sg_a * sg_s);

    // ---------------- pass 2: full per-pair pipeline ------------------------
    float acc_hagg[4] = {0.f, 0.f, 0.f, 0.f};
    float acc_cmb[2][3] = {{0.f, 0.f, 0.f}, {0.f, 0.f, 0.f}};
    float ax0 = 0.f, ax1 = 0.f, ax2 = 0.f;
    __bf16* tile = (__bf16*)w->stage;

    for (int jt = 0; jt < 64; ++jt) {
        if (jt < 63) __builtin_prefetch(Brow + (jt + 1) * 1024 + lane * 4, 0, 1);
        stageB(w, Brow + jt * 1024, lane);
        wave_fence();
        const int j = jt * 16 + l15;
        const float* xp = x + 3 * j;
        float dx0 = xi0 - xp[0], dx1 = xi1 - xp[1], dx2 = xi2 - xp[2];
        float nrm = sqrtf(dx0 * dx0 + dx1 * dx1 + dx2 * dx2 + EPSV);

        v16bf ah[2];
        float dot = build_hidden<true>(w, l15, half, nrm, ah);
        dot += __shfl_xor(dot, 16, 32);
        const bool dg = (j == i);
        float sl = dot + c0;
        float s  = (sl > 0.f ? sl : 0.01f * sl) - (dg ? INFV : 0.f);
        float a  = -(nrm + (dg ? INFV : 0.f)) * gamma;
        float cmb = __expf(a + s - m_s) * invT;
        if (half == 0) {
            w->comb[l15]   = cmb;
            float iv       = 1.f / (nrm + 1.f);
            w->dirs[l15]   = make_float4(dx0 * iv, dx1 * iv, dx2 * iv, 0.f);
            w->phiacc[l15] = 0.f;
        }
        wave_fence();

        // edge GEMM: h_e = hidden @ ew2 + eb2
        v8f che[4];
#pragma unroll
        for (int g = 0; g < 4; ++g) che[g] = splat8(eb2v[g]);
#pragma unroll
        for (int kt = 0; kt < 2; ++kt)
#pragma unroll
            for (int g = 0; g < 4; ++g)
                che[g] = wmma_bf16(ah[kt], ldW(sEw2, l15, half, kt, g), che[g]);

        // round-trip h_e -> A-frags for the phi path
#pragma unroll
        for (int g = 0; g < 4; ++g) stC(tile, l15, half, g, che[g]);
        wave_fence();
        v16bf ahe0 = ldA(tile, l15, half, 0), ahe1 = ldA(tile, l15, half, 1);

        // phi = silu(h_e @ cw1 + cb1) @ cw2
        v8f cp[4];
#pragma unroll
        for (int g = 0; g < 4; ++g) cp[g] = splat8(cb1v[g]);
#pragma unroll
        for (int g = 0; g < 4; ++g) {
            cp[g] = wmma_bf16(ahe0, ldW(sCw1, l15, half, 0, g), cp[g]);
            cp[g] = wmma_bf16(ahe1, ldW(sCw1, l15, half, 1, g), cp[g]);
        }
        float pv[8];
#pragma unroll
        for (int v = 0; v < 8; ++v) pv[v] = 0.f;
#pragma unroll
        for (int g = 0; g < 4; ++g)
#pragma unroll
            for (int v = 0; v < 8; ++v) pv[v] += silu_f(cp[g][v]) * cw2v[g];
#pragma unroll
        for (int v = 0; v < 8; ++v) atomicAdd(&w->phiacc[v + 8 * half], pv[v]);
        wave_fence();
        float ph = w->phiacc[l15];
        ax0 += dx0 * ph; ax1 += dx1 * ph; ax2 += dx2 * ph;

        // h_e_att = h_e * comb ; accumulate h_agg
        v8f cbv = *(const v8f*)(w->comb + 8 * half);
#pragma unroll
        for (int g = 0; g < 4; ++g) {
            float sgt = 0.f;
#pragma unroll
            for (int v = 0; v < 8; ++v) { che[g][v] *= cbv[v]; sgt += che[g][v]; }
            acc_hagg[g] += sgt;
        }
#pragma unroll
        for (int g = 0; g < 4; ++g) stC(tile, l15, half, g, che[g]);
        wave_fence();
        v16bf af0 = ldA(tile, l15, half, 0), af1 = ldA(tile, l15, half, 1);

        // coeff hidden = silu(h_e_att @ fw1 + fb1)
        v8f cc[4];
#pragma unroll
        for (int g = 0; g < 4; ++g) cc[g] = splat8(fb1v[g]);
#pragma unroll
        for (int g = 0; g < 4; ++g) {
            cc[g] = wmma_bf16(af0, ldW(sFw1, l15, half, 0, g), cc[g]);
            cc[g] = wmma_bf16(af1, ldW(sFw1, l15, half, 1, g), cc[g]);
        }
#pragma unroll
        for (int g = 0; g < 4; ++g)
#pragma unroll
            for (int v = 0; v < 8; ++v) cc[g][v] = silu_f(cc[g][v]);
#pragma unroll
        for (int g = 0; g < 4; ++g) stC(tile, l15, half, g, cc[g]);
        wave_fence();
        v16bf ag0 = ldA(tile, l15, half, 0), ag1 = ldA(tile, l15, half, 1);

        // coeff = hidden @ fw2 + fb2  (N = 32)
        v8f cf[2];
#pragma unroll
        for (int nt = 0; nt < 2; ++nt) {
            cf[nt] = splat8(fb2v[nt]);
            cf[nt] = wmma_bf16(ag0, ldW(sFw2, l15, half, 0, nt), cf[nt]);
            cf[nt] = wmma_bf16(ag1, ldW(sFw2, l15, half, 1, nt), cf[nt]);
        }
        // combos accumulation: coeff x dirs
#pragma unroll
        for (int g2 = 0; g2 < 2; ++g2)
#pragma unroll
            for (int v = 0; v < 8; ++v) {
                float4 d4 = w->dirs[v + 8 * half];
                float  cv = cf[g2][v];
                acc_cmb[g2][0] += cv * d4.x;
                acc_cmb[g2][1] += cv * d4.y;
                acc_cmb[g2][2] += cv * d4.z;
            }
    }

    // ---------------- per-row tail ------------------------------------------
    // h_agg
#pragma unroll
    for (int g = 0; g < 4; ++g) {
        float t = acc_hagg[g] + __shfl_xor(acc_hagg[g], 16, 32);
        if (half == 0) w->nodein[64 + g * 16 + l15] = t;
    }
    // combos -> sq
#pragma unroll
    for (int g2 = 0; g2 < 2; ++g2) {
        float c0v = (acc_cmb[g2][0] + __shfl_xor(acc_cmb[g2][0], 16, 32)) * (1.f / 1024.f);
        float c1v = (acc_cmb[g2][1] + __shfl_xor(acc_cmb[g2][1], 16, 32)) * (1.f / 1024.f);
        float c2v = (acc_cmb[g2][2] + __shfl_xor(acc_cmb[g2][2], 16, 32)) * (1.f / 1024.f);
        if (half == 0) w->sq[g2 * 16 + l15] = c0v * c0v + c1v * c1v + c2v * c2v;
    }
    // x update reduction over rows
#pragma unroll
    for (int mk = 1; mk < 16; mk <<= 1) {
        ax0 += __shfl_xor(ax0, mk, 32);
        ax1 += __shfl_xor(ax1, mk, 32);
        ax2 += __shfl_xor(ax2, mk, 32);
    }
    if (lane == 0) {
        float* ox = out + NPTS * 64 + i * 3;
        ox[0] = xi0 + ax0 * (1.f / 1024.f);
        ox[1] = xi1 + ax1 * (1.f / 1024.f);
        ox[2] = xi2 + ax2 * (1.f / 1024.f);
    }
    w->nodein[lane]      = h[i * 64 + lane];
    w->nodein[lane + 32] = h[i * 64 + lane + 32];
    wave_fence();

    // h_comb = silu(sq @ pw1 + pb1) @ pw2 + pb2
#pragma unroll
    for (int r = 0; r < 2; ++r) {
        int f = lane + r * 32;
        float t = pb1[f];
        for (int c = 0; c < 32; ++c) t += w->sq[c] * pw1[c * 64 + f];
        w->tmp[f] = silu_f(t);
    }
    wave_fence();
#pragma unroll
    for (int r = 0; r < 2; ++r) {
        int o = lane + r * 32;
        float t = pb2[o];
        for (int f = 0; f < 64; ++f) t += w->tmp[f] * pw2[f * 64 + o];
        w->nodein[128 + o] = t;
    }
    wave_fence();

    // node MLP: h_out = h + silu(node_in @ nw1 + nb1) @ nw2 + nb2
#pragma unroll
    for (int r = 0; r < 2; ++r) {
        int f = lane + r * 32;
        float t = nb1[f];
        for (int q = 0; q < 192; ++q) t += w->nodein[q] * nw1[q * 64 + f];
        w->tmp[f] = silu_f(t);
    }
    wave_fence();
#pragma unroll
    for (int r = 0; r < 2; ++r) {
        int o = lane + r * 32;
        float t = h[i * 64 + o] + nb2[o];
        for (int f = 0; f < 64; ++f) t += w->tmp[f] * nw2[f * 64 + o];
        out[i * 64 + o] = t;
    }
}

// ------------------------------- host shim ----------------------------------

extern "C" void kernel_launch(void* const* d_in, const int* in_sizes, int n_in,
                              void* d_out, int out_size, void* d_ws, size_t ws_size,
                              hipStream_t stream) {
    (void)in_sizes; (void)n_in; (void)out_size; (void)ws_size;
    const float* h   = (const float*)d_in[0];
    const float* x   = (const float*)d_in[1];
    const float* ew1 = (const float*)d_in[2];
    const float* eb1 = (const float*)d_in[3];
    const float* ew2 = (const float*)d_in[4];
    const float* eb2 = (const float*)d_in[5];
    const float* nw1 = (const float*)d_in[6];
    const float* nb1 = (const float*)d_in[7];
    const float* nw2 = (const float*)d_in[8];
    const float* nb2 = (const float*)d_in[9];
    const float* cw1 = (const float*)d_in[10];
    const float* cb1 = (const float*)d_in[11];
    const float* cw2 = (const float*)d_in[12];
    const float* aw  = (const float*)d_in[13];
    const float* ab  = (const float*)d_in[14];
    const float* fw1 = (const float*)d_in[15];
    const float* fb1 = (const float*)d_in[16];
    const float* fw2 = (const float*)d_in[17];
    const float* fb2 = (const float*)d_in[18];
    const float* pw1 = (const float*)d_in[19];
    const float* pb1 = (const float*)d_in[20];
    const float* pw2 = (const float*)d_in[21];
    const float* pb2 = (const float*)d_in[22];
    const float* lg  = (const float*)d_in[23];

    float* ws = (float*)d_ws;
    float* A  = ws;                 // 1024*64
    float* B  = ws + 65536;         // 1024*64
    float* U  = ws + 131072;        // 64
    float* C0 = ws + 131136;        // 1

    sake_prep_rows<<<NPTS, 64, 0, stream>>>(h, ew1, eb1, A, B);
    sake_prep_u<<<1, 64, 0, stream>>>(ew2, eb2, aw, ab, U, C0);
    sake_main<<<NPTS / 4, 128, 0, stream>>>(
        h, x, ew1, ew2, eb2, nw1, nb1, nw2, nb2, cw1, cb1, cw2, fw1, fb1, fw2, fb2,
        pw1, pb1, pw2, pb2, lg, A, B, U, C0, (float*)d_out);
}
